// GraphCastProcessor_69621419868957
// MI455X (gfx1250) — compile-verified, matrix operand order checked by reading, not run on hardware
//
#include <hip/hip_runtime.h>

typedef unsigned short u16;
typedef unsigned int   u32;

typedef __attribute__((ext_vector_type(16))) __bf16 v16bf;
typedef __attribute__((ext_vector_type(8)))  float  v8f;

#define NNODES 40962
#define NEDGES 327680
#define ND 256
#define ED 64
#define HD 512
#define NLAY 16
#define EIN 576   // ED + 2*ND
#define NIN 320   // ND + ED
#define LN_EPS 1e-5f

// Fragment sizes: one 16x16x32 bf16 B-fragment = 32 lanes * 16 elems = 512 u16
#define FRAG_U16 512

union FragAB {
    v16bf bf;
    u32   u[8];
    u16   us[16];
};

__device__ __forceinline__ u16 f2bf(float f) {
    u32 u = __float_as_uint(f);
    u32 r = (u + 0x7fffu + ((u >> 16) & 1u)) >> 16;   // round-to-nearest-even
    return (u16)r;
}

__device__ __forceinline__ float silu(float v) {
    return v * __frcp_rn(1.0f + __expf(-v));
}

// Element i of a 16-bit A/B fragment in lane `lane` holds K index:
//   K = (i&7) + ((i>>3)<<4) + ((lane>>4)<<3)      (ISA 7.12.2, 16-bit 16x32 layout)

// Load B-fragment from pre-swizzled global weights (fragment-major).
__device__ __forceinline__ v16bf load_bfrag(const u16* __restrict__ frag, int lane) {
    FragAB f;
    const uint4* p = (const uint4*)(frag + lane * 16);
    uint4 a = p[0], b = p[1];
    f.u[0] = a.x; f.u[1] = a.y; f.u[2] = a.z; f.u[3] = a.w;
    f.u[4] = b.x; f.u[5] = b.y; f.u[6] = b.z; f.u[7] = b.w;
    return f.bf;
}

// Load A-fragment from LDS tile (row-major bf16, row stride RS u16s).
__device__ __forceinline__ v16bf load_afrag(const u16* __restrict__ lds, int row0,
                                            int RS, int kbase, int lane) {
    FragAB f;
    int m = row0 + (lane & 15);
    int kh = (lane >> 4) << 3;
    const u32* rowp = (const u32*)(lds + (size_t)m * RS);
#pragma unroll
    for (int t = 0; t < 8; ++t) {
        int i = 2 * t;
        int k = kbase + (i & 7) + ((i >> 3) << 4) + kh;  // even -> aligned u32 pair (k, k+1)
        f.u[t] = rowp[k >> 1];
    }
    return f.bf;
}

// ---------------------------------------------------------------------------
// Weight swizzle: f32 [L][K][N] row-major -> bf16 fragment-major
// grid.x = L*KT*NT blocks of 32 threads (one wave builds one fragment)
// ---------------------------------------------------------------------------
__global__ void swizzle_weights(const float* __restrict__ W, u16* __restrict__ out,
                                int K, int Nn, int KT, int NT) {
    int f    = blockIdx.x;
    int nt   = f % NT;
    int kt   = (f / NT) % KT;
    int l    = f / (NT * KT);
    int lane = threadIdx.x;
    const float* Wl = W + (size_t)l * K * Nn;
    u16* o = out + (size_t)f * FRAG_U16 + lane * 16;
    int col = nt * 16 + (lane & 15);
    int kh  = (lane >> 4) << 3;
#pragma unroll
    for (int i = 0; i < 16; ++i) {
        int k = kt * 32 + (i & 7) + ((i >> 3) << 4) + kh;
        o[i] = f2bf(Wl[(size_t)k * Nn + col]);
    }
}

// ---------------------------------------------------------------------------
// Edge update kernel: 32 edges per block, 256 threads (8 waves).
//   m = [e | x[src] | x[dst]]  (32 x 576 bf16 in LDS)
//   h = silu(m @ We1 + be1)    (32 x 512 bf16 in LDS)
//   o = h @ We2 + be2          (32 x 64 f32 in LDS)
//   e_next = e + LN(o)*g + b ; atomicAdd into agg[dst]
// ---------------------------------------------------------------------------
__global__ __launch_bounds__(256, 2)
void edge_kernel(const float* __restrict__ e_in, const float* __restrict__ x_in,
                 const int* __restrict__ eidx,
                 const u16* __restrict__ we1s, const float* __restrict__ be1,
                 const u16* __restrict__ we2s, const float* __restrict__ be2,
                 const float* __restrict__ g_e, const float* __restrict__ b_e,
                 float* __restrict__ e_out, float* __restrict__ agg) {
    __shared__ u16  smem[32 * EIN];      // 36 KB; reused for h (32x512) and out (32x64 f32)
    __shared__ int  s_src[32], s_dst[32];
    __shared__ float s_mean[32], s_rstd[32];

    const int tid  = threadIdx.x;
    const int lane = tid & 31;
    const int wave = tid >> 5;
    const long e0  = (long)blockIdx.x * 32;

    if (tid < 32) {
        s_src[tid] = eidx[e0 + tid];
        s_dst[tid] = eidx[NEDGES + e0 + tid];
    }
    __syncthreads();

    // ---- gather phase: 8 threads per edge row, bf16 into LDS -------------
    {
        const int r = tid >> 3, u = tid & 7;
        const float* erow = e_in + (e0 + r) * ED;
        const float* xs   = x_in + (size_t)s_src[r] * ND;
        const float* xd   = x_in + (size_t)s_dst[r] * ND;
        u16* mrow = smem + r * EIN;
#pragma unroll
        for (int kk = 0; kk < EIN / 32; ++kk) {
            int c = u * 4 + kk * 32;
            float4 v;
            if (c < ED)            v = *(const float4*)(erow + c);
            else if (c < ED + ND)  v = *(const float4*)(xs + (c - ED));
            else                   v = *(const float4*)(xd + (c - ED - ND));
            u32 p0 = (u32)f2bf(v.x) | ((u32)f2bf(v.y) << 16);
            u32 p1 = (u32)f2bf(v.z) | ((u32)f2bf(v.w) << 16);
            *(uint2*)(mrow + c) = make_uint2(p0, p1);
        }
    }
    __syncthreads();

    // ---- GEMM1: (32 x 576) @ (576 x 512), wave owns 64 output columns ----
    const v8f vzero = {0.f, 0.f, 0.f, 0.f, 0.f, 0.f, 0.f, 0.f};
    v8f acc[2][4];
#pragma unroll
    for (int ms = 0; ms < 2; ++ms)
#pragma unroll
        for (int ns = 0; ns < 4; ++ns) acc[ms][ns] = vzero;

    const int nb = wave * 4;                       // first N-subtile for this wave
    for (int kt = 0; kt < EIN / 32; ++kt) {
        v16bf a0 = load_afrag(smem, 0,  EIN, kt * 32, lane);
        v16bf a1 = load_afrag(smem, 16, EIN, kt * 32, lane);
#pragma unroll
        for (int ns = 0; ns < 4; ++ns) {
            const u16* bp = we1s + ((size_t)(kt * (HD / 16) + nb + ns)) * FRAG_U16;
            v16bf b = load_bfrag(bp, lane);
            acc[0][ns] = __builtin_amdgcn_wmma_f32_16x16x32_bf16(
                false, a0, false, b, (short)0, acc[0][ns], false, false);
            acc[1][ns] = __builtin_amdgcn_wmma_f32_16x16x32_bf16(
                false, a1, false, b, (short)0, acc[1][ns], false, false);
        }
    }
    __syncthreads();                                // done reading m-tile

    // ---- SiLU epilogue -> h (bf16, row stride 512) in LDS ----------------
    {
        const int colb = lane & 15;
        const int kh   = (lane >> 4) * 8;
#pragma unroll
        for (int ms = 0; ms < 2; ++ms)
#pragma unroll
            for (int ns = 0; ns < 4; ++ns) {
                int col = (nb + ns) * 16 + colb;
                float bias = be1[col];
#pragma unroll
                for (int j = 0; j < 8; ++j) {
                    int row = ms * 16 + kh + j;
                    smem[row * HD + col] = f2bf(silu(acc[ms][ns][j] + bias));
                }
            }
    }
    __syncthreads();

    // ---- GEMM2: (32 x 512) @ (512 x 64): one 16x16 tile per wave ---------
    const int ms2 = wave & 1;
    const int ns2 = wave >> 1;                      // 0..3
    v8f acc2 = vzero;
    for (int kt = 0; kt < HD / 32; ++kt) {
        v16bf a = load_afrag(smem, ms2 * 16, HD, kt * 32, lane);
        const u16* bp = we2s + ((size_t)(kt * (ED / 16) + ns2)) * FRAG_U16;
        v16bf b = load_bfrag(bp, lane);
        acc2 = __builtin_amdgcn_wmma_f32_16x16x32_bf16(
            false, a, false, b, (short)0, acc2, false, false);
    }
    __syncthreads();                                // done reading h-tile

    // ---- stage f32 output (32 x 64) ----
    float* outs = (float*)smem;
    {
        const int col = ns2 * 16 + (lane & 15);
        const int kh  = (lane >> 4) * 8;
        float bias = be2[col];
#pragma unroll
        for (int j = 0; j < 8; ++j)
            outs[(ms2 * 16 + kh + j) * ED + col] = acc2[j] + bias;
    }
    __syncthreads();

    // ---- LayerNorm stats over 64 cols ----
    if (tid < 32) {
        const float* r = outs + tid * ED;
        float s = 0.f, s2 = 0.f;
        for (int c = 0; c < ED; ++c) { float v = r[c]; s += v; s2 += v * v; }
        float m = s * (1.f / ED);
        s_mean[tid] = m;
        s_rstd[tid] = __frsqrt_rn(s2 * (1.f / ED) - m * m + LN_EPS);
    }
    __syncthreads();

    // ---- apply LN + residual, store e_next, scatter into agg -------------
    {
        const int r = tid >> 3, u = tid & 7;
        const long edge = e0 + r;
        const int  d    = s_dst[r];
        const float m = s_mean[r], rs = s_rstd[r];
#pragma unroll
        for (int q = 0; q < 8; ++q) {
            int c = u * 8 + q;
            float v = (outs[r * ED + c] - m) * rs * g_e[c] + b_e[c];
            float o = e_in[edge * ED + c] + v;
            e_out[edge * ED + c] = o;
            atomicAdd(&agg[(size_t)d * ED + c], o);
        }
    }
}

// ---------------------------------------------------------------------------
// Node update kernel: 32 nodes per block, 256 threads (8 waves).
//   m = [x | agg]  (32 x 320)
//   h = silu(m @ Wn1 + bn1)   (32 x 512)
//   o = h @ Wn2 + bn2         (32 x 256)
//   x_next = x + LN(o)*g + b
// ---------------------------------------------------------------------------
__global__ __launch_bounds__(256, 2)
void node_kernel(const float* __restrict__ x_in, const float* __restrict__ agg,
                 const u16* __restrict__ wn1s, const float* __restrict__ bn1,
                 const u16* __restrict__ wn2s, const float* __restrict__ bn2,
                 const float* __restrict__ g_n, const float* __restrict__ b_n,
                 float* __restrict__ x_out) {
    __shared__ u16   smem[32 * HD];     // 32 KB; m (32x320), then h (32x512), then out f32 (32x256)
    __shared__ float s_mean[32], s_rstd[32];

    const int tid  = threadIdx.x;
    const int lane = tid & 31;
    const int wave = tid >> 5;
    const long n0  = (long)blockIdx.x * 32;

    // ---- gather [x | agg] -> bf16 LDS ----
    {
        const int r = tid >> 3, u = tid & 7;
        long n = n0 + r; if (n >= NNODES) n = NNODES - 1;
        const float* xr = x_in + (size_t)n * ND;
        const float* ar = agg + (size_t)n * ED;
        u16* mrow = smem + r * NIN;
#pragma unroll
        for (int kk = 0; kk < NIN / 32; ++kk) {
            int c = u * 4 + kk * 32;
            float4 v;
            if (c < ND) v = *(const float4*)(xr + c);
            else        v = *(const float4*)(ar + (c - ND));
            u32 p0 = (u32)f2bf(v.x) | ((u32)f2bf(v.y) << 16);
            u32 p1 = (u32)f2bf(v.z) | ((u32)f2bf(v.w) << 16);
            *(uint2*)(mrow + c) = make_uint2(p0, p1);
        }
    }
    __syncthreads();

    // ---- GEMM1: (32 x 320) @ (320 x 512) ----
    const v8f vzero = {0.f, 0.f, 0.f, 0.f, 0.f, 0.f, 0.f, 0.f};
    v8f acc[2][4];
#pragma unroll
    for (int ms = 0; ms < 2; ++ms)
#pragma unroll
        for (int ns = 0; ns < 4; ++ns) acc[ms][ns] = vzero;

    const int nb = wave * 4;
    for (int kt = 0; kt < NIN / 32; ++kt) {
        v16bf a0 = load_afrag(smem, 0,  NIN, kt * 32, lane);
        v16bf a1 = load_afrag(smem, 16, NIN, kt * 32, lane);
#pragma unroll
        for (int ns = 0; ns < 4; ++ns) {
            const u16* bp = wn1s + ((size_t)(kt * (HD / 16) + nb + ns)) * FRAG_U16;
            v16bf b = load_bfrag(bp, lane);
            acc[0][ns] = __builtin_amdgcn_wmma_f32_16x16x32_bf16(
                false, a0, false, b, (short)0, acc[0][ns], false, false);
            acc[1][ns] = __builtin_amdgcn_wmma_f32_16x16x32_bf16(
                false, a1, false, b, (short)0, acc[1][ns], false, false);
        }
    }
    __syncthreads();

    // ---- SiLU -> h (bf16, stride 512) ----
    {
        const int colb = lane & 15;
        const int kh   = (lane >> 4) * 8;
#pragma unroll
        for (int ms = 0; ms < 2; ++ms)
#pragma unroll
            for (int ns = 0; ns < 4; ++ns) {
                int col = (nb + ns) * 16 + colb;
                float bias = bn1[col];
#pragma unroll
                for (int j = 0; j < 8; ++j) {
                    int row = ms * 16 + kh + j;
                    smem[row * HD + col] = f2bf(silu(acc[ms][ns][j] + bias));
                }
            }
    }
    __syncthreads();

    // ---- GEMM2: (32 x 512) @ (512 x 256): 4 tiles per wave ---------------
    const int ms2 = wave & 1;
    const int nb2 = (wave >> 1) * 4;                // 0,4,8,12
    v8f acc2[4];
#pragma unroll
    for (int ns = 0; ns < 4; ++ns) acc2[ns] = vzero;

    for (int kt = 0; kt < HD / 32; ++kt) {
        v16bf a = load_afrag(smem, ms2 * 16, HD, kt * 32, lane);
#pragma unroll
        for (int ns = 0; ns < 4; ++ns) {
            const u16* bp = wn2s + ((size_t)(kt * (ND / 16) + nb2 + ns)) * FRAG_U16;
            v16bf b = load_bfrag(bp, lane);
            acc2[ns] = __builtin_amdgcn_wmma_f32_16x16x32_bf16(
                false, a, false, b, (short)0, acc2[ns], false, false);
        }
    }
    __syncthreads();

    // ---- stage f32 output (32 x 256) ----
    float* outs = (float*)smem;
    {
        const int kh = (lane >> 4) * 8;
#pragma unroll
        for (int ns = 0; ns < 4; ++ns) {
            int col = (nb2 + ns) * 16 + (lane & 15);
            float bias = bn2[col];
#pragma unroll
            for (int j = 0; j < 8; ++j)
                outs[(ms2 * 16 + kh + j) * ND + col] = acc2[ns][j] + bias;
        }
    }
    __syncthreads();

    // ---- LayerNorm over 256 cols ----
    if (tid < 32) {
        const float* r = outs + tid * ND;
        float s = 0.f, s2 = 0.f;
        for (int c = 0; c < ND; ++c) { float v = r[c]; s += v; s2 += v * v; }
        float m = s * (1.f / ND);
        s_mean[tid] = m;
        s_rstd[tid] = __frsqrt_rn(s2 * (1.f / ND) - m * m + LN_EPS);
    }
    __syncthreads();

    // ---- apply LN + residual, store x_next ----
    {
        const int r = tid >> 3, u = tid & 7;
        long n = n0 + r;
        if (n < NNODES) {
            const float m = s_mean[r], rs = s_rstd[r];
#pragma unroll
            for (int q = 0; q < 32; ++q) {
                int c = u * 32 + q;
                float v = (outs[r * ND + c] - m) * rs * g_n[c] + b_n[c];
                x_out[(size_t)n * ND + c] = x_in[(size_t)n * ND + c] + v;
            }
        }
    }
}

// ---------------------------------------------------------------------------
extern "C" void kernel_launch(void* const* d_in, const int* in_sizes, int n_in,
                              void* d_out, int out_size, void* d_ws, size_t ws_size,
                              hipStream_t stream) {
    const float* node_features = (const float*)d_in[0];
    const float* edge_features = (const float*)d_in[1];
    const float* We1 = (const float*)d_in[2];
    const float* be1 = (const float*)d_in[3];
    const float* We2 = (const float*)d_in[4];
    const float* be2 = (const float*)d_in[5];
    const float* ge  = (const float*)d_in[6];
    const float* ble = (const float*)d_in[7];
    const float* Wn1 = (const float*)d_in[8];
    const float* bn1 = (const float*)d_in[9];
    const float* Wn2 = (const float*)d_in[10];
    const float* bn2 = (const float*)d_in[11];
    const float* gn  = (const float*)d_in[12];
    const float* bln = (const float*)d_in[13];
    const int* edge_index = (const int*)d_in[14];
    float* out = (float*)d_out;

    // Fragment counts per layer
    const size_t F_WE1 = (EIN / 32) * (HD / 16);   // 18*32 = 576
    const size_t F_WE2 = (HD  / 32) * (ED / 16);   // 16*4  = 64
    const size_t F_WN1 = (NIN / 32) * (HD / 16);   // 10*32 = 320
    const size_t F_WN2 = (HD  / 32) * (ND / 16);   // 16*16 = 256

    char* ws = (char*)d_ws;
    size_t off = 0;
    auto take = [&](size_t bytes) { char* p = ws + off; off = (off + bytes + 255) & ~(size_t)255; return p; };

    u16* we1s = (u16*)take(NLAY * F_WE1 * FRAG_U16 * 2);
    u16* we2s = (u16*)take(NLAY * F_WE2 * FRAG_U16 * 2);
    u16* wn1s = (u16*)take(NLAY * F_WN1 * FRAG_U16 * 2);
    u16* wn2s = (u16*)take(NLAY * F_WN2 * FRAG_U16 * 2);

    const size_t EB = (size_t)NEDGES * ED * 4;
    const size_t XB = (size_t)NNODES * ND * 4;
    const size_t AB = (size_t)NNODES * ED * 4;
    float* ebuf[2] = {(float*)take(EB), (float*)take(EB)};
    float* xbuf[2] = {(float*)take(XB), (float*)take(XB)};
    float* agg = (float*)take(AB);

    // ---- swizzle weights to WMMA fragment layout (bf16) ----
    swizzle_weights<<<NLAY * (int)F_WE1, 32, 0, stream>>>(We1, we1s, EIN, HD, EIN / 32, HD / 16);
    swizzle_weights<<<NLAY * (int)F_WE2, 32, 0, stream>>>(We2, we2s, HD, ED, HD / 32, ED / 16);
    swizzle_weights<<<NLAY * (int)F_WN1, 32, 0, stream>>>(Wn1, wn1s, NIN, HD, NIN / 32, HD / 16);
    swizzle_weights<<<NLAY * (int)F_WN2, 32, 0, stream>>>(Wn2, wn2s, HD, ND, HD / 32, ND / 16);

    const int edge_blocks = NEDGES / 32;
    const int node_blocks = (NNODES + 31) / 32;

    for (int b = 0; b < 2; ++b) {
        hipMemcpyAsync(xbuf[0], node_features + (size_t)b * NNODES * ND, XB,
                       hipMemcpyDeviceToDevice, stream);
        hipMemcpyAsync(ebuf[0], edge_features, EB, hipMemcpyDeviceToDevice, stream);
        int cur = 0;
        for (int l = 0; l < NLAY; ++l) {
            hipMemsetAsync(agg, 0, AB, stream);
            edge_kernel<<<edge_blocks, 256, 0, stream>>>(
                ebuf[cur], xbuf[cur], edge_index,
                we1s + (size_t)l * F_WE1 * FRAG_U16, be1 + (size_t)l * HD,
                we2s + (size_t)l * F_WE2 * FRAG_U16, be2 + (size_t)l * ED,
                ge + (size_t)l * ED, ble + (size_t)l * ED,
                ebuf[1 - cur], agg);
            node_kernel<<<node_blocks, 256, 0, stream>>>(
                xbuf[cur], agg,
                wn1s + (size_t)l * F_WN1 * FRAG_U16, bn1 + (size_t)l * HD,
                wn2s + (size_t)l * F_WN2 * FRAG_U16, bn2 + (size_t)l * ND,
                gn + (size_t)l * ND, bln + (size_t)l * ND,
                xbuf[1 - cur]);
            cur ^= 1;
        }
        hipMemcpyAsync(out + (size_t)b * NNODES * ND, xbuf[cur], XB,
                       hipMemcpyDeviceToDevice, stream);
    }
}
